// gumbel_shGLM_41910290874961
// MI455X (gfx1250) — matrix operand order repaired and checked by reading, not compiled, not run on hardware
//
#include <hip/hip_runtime.h>
#include <math.h>

typedef __attribute__((ext_vector_type(16))) _Float16 v16h;
typedef __attribute__((ext_vector_type(8)))  float    v8f;
typedef __attribute__((ext_vector_type(8)))  _Float16 v8h;
typedef __attribute__((ext_vector_type(4)))  float    f32x4;

#define T_DATA 20000
#define E_NO   2000
#define I_NO   500
#define SUB    12
#define T_SYN  200
#define SB     3
#define CONV_T 32
#define KPAD_E 2048
#define KPAD_I 512
#define CONV_P 13   // padded row stride (odd -> LDS bank-conflict free)

// ---------------------------------------------------------------------------
// Kernel 1: synaptic kernels  k[s][c][t] = sum_b W[s][b][c] * tt*exp(-tt)
// ---------------------------------------------------------------------------
__global__ void syn_kernels_k(const float* __restrict__ W,      // [SUB][SB][2]
                              const float* __restrict__ Delta,  // [SUB][2]
                              float* __restrict__ kern)         // [SUB][2][T_SYN]
{
    int idx = blockIdx.x * blockDim.x + threadIdx.x;
    if (idx >= SUB * 2 * T_SYN) return;
    int t = idx % T_SYN;
    int c = (idx / T_SYN) % 2;
    int s = idx / (2 * T_SYN);
    float ts  = fmaxf((float)t - expf(Delta[s * 2 + c]), 0.0f);
    float acc = 0.0f;
#pragma unroll
    for (int b = 0; b < SB; ++b) {
        float tau = expf(0.5f * (float)b);
        float tt  = ts / tau;
        acc += W[(s * SB + b) * 2 + c] * (tt * expf(-tt));
    }
    kern[idx] = acc;
}

// ---------------------------------------------------------------------------
// Kernel 1b: pack C_syn into padded fp16 B matrix [16][Kpad], zero-padded.
// Rows 12..15 and K >= Ktot are zero, so the GEMM B loads need NO guards.
// ---------------------------------------------------------------------------
__global__ void pack_b_k(const float* __restrict__ C, _Float16* __restrict__ B,
                         int Ktot, int Kpad)
{
    int idx = blockIdx.x * blockDim.x + threadIdx.x;
    if (idx >= 16 * Kpad) return;
    int n = idx / Kpad, K = idx % Kpad;
    float v = 0.0f;
    if (n < SUB && K < Ktot) v = C[n * Ktot + K];
    B[idx] = (_Float16)v;
}

// ---------------------------------------------------------------------------
// A-fragment loaders (ISA 16-bit 16x32 A layout; lane = M row,
// lanes 0-15: K = kc+{0..7, 16..23}; lanes 16-31: K = kc+{8..15, 24..31})
// ---------------------------------------------------------------------------
__device__ __forceinline__ v16h load_a_full(const float* __restrict__ base,
                                            int kc, int aoff)
{
    const f32x4* p0 = (const f32x4*)(base + kc + aoff);
    const f32x4* p1 = (const f32x4*)(base + kc + 16 + aoff);
    f32x4 q0 = p0[0], q1 = p0[1], r0 = p1[0], r1 = p1[1];
    v16h a;
#pragma unroll
    for (int e = 0; e < 4; ++e) {
        a[e]      = (_Float16)q0[e];
        a[4 + e]  = (_Float16)q1[e];
        a[8 + e]  = (_Float16)r0[e];
        a[12 + e] = (_Float16)r1[e];
    }
    return a;
}

// E tail: 2000 % 32 == 16 -> remaining K fills exactly elements 0..7 of both
// lane halves; elements 8..15 are zero. Fully unconditional.
__device__ __forceinline__ v16h load_a_tailE(const float* __restrict__ base,
                                             int kc, int aoff)
{
    const f32x4* p0 = (const f32x4*)(base + kc + aoff);
    f32x4 q0 = p0[0], q1 = p0[1];
    v16h a = {};
#pragma unroll
    for (int e = 0; e < 4; ++e) {
        a[e]     = (_Float16)q0[e];
        a[4 + e] = (_Float16)q1[e];
    }
    return a;
}

// I tail: 500 % 32 == 20 -> elements 0..7 valid for both halves; elements
// 8..15 partially valid. Address-clamp + select: loads stay unconditional
// (v_cndmask, no EXEC divergence).
__device__ __forceinline__ v16h load_a_tailI(const float* __restrict__ base,
                                             int kc, int aoff)
{
    const f32x4* p0 = (const f32x4*)(base + kc + aoff);
    f32x4 q0 = p0[0], q1 = p0[1];
    v16h a;
#pragma unroll
    for (int e = 0; e < 4; ++e) {
        a[e]     = (_Float16)q0[e];
        a[4 + e] = (_Float16)q1[e];
    }
#pragma unroll
    for (int e = 8; e < 16; ++e) {
        int K = kc + 16 + aoff + (e - 8);
        bool ok = K < I_NO;
        int Kc = ok ? K : 0;
        float v = base[Kc];
        a[e] = ok ? (_Float16)v : (_Float16)0.0f;
    }
    return a;
}

// ---------------------------------------------------------------------------
// Kernel 2: spike GEMMs via WMMA f16 (spikes are exact 0/1 -> f16 exact).
// 2 M-tiles (32 T rows) per wave; B fragment = two unconditional 16B loads
// per lane from the padded fp16 B (L2-resident).
// ---------------------------------------------------------------------------
__global__ __launch_bounds__(128) void spike_gemm_k(
    const float* __restrict__ S_e, const float* __restrict__ S_i,
    const _Float16* __restrict__ Be, const _Float16* __restrict__ Bi,
    float* __restrict__ IN_e, float* __restrict__ IN_i)
{
    const int lane  = threadIdx.x & 31;
    const int wave  = threadIdx.x >> 5;
    const int pair  = blockIdx.x * 4 + wave;       // wave-uniform
    const int tile0 = pair * 2;
    if (tile0 * 16 >= T_DATA) return;              // EXEC stays all-ones inside

    const int row  = lane & 15;
    const int hi   = (lane >> 4) & 1;
    const int aoff = hi ? 8 : 0;
    const int boff = hi ? 8 : 0;

    const size_t gRow0 = (size_t)(tile0 * 16 + row);
    const size_t gRow1 = gRow0 + 16;

    v8f accE0 = {}, accE1 = {}, accI0 = {}, accI1 = {};

    // ---------------- E path: K = 2000 ----------------
    {
        const float* b0 = S_e + gRow0 * E_NO;
        const float* b1 = S_e + gRow1 * E_NO;
        const _Float16* bp = Be + (size_t)row * KPAD_E + boff;
        int kc = 0;
        for (; kc + 32 <= E_NO; kc += 32) {
            v16h a0 = load_a_full(b0, kc, aoff);
            v16h a1 = load_a_full(b1, kc, aoff);
            v16h b  = __builtin_shufflevector(*(const v8h*)(bp + kc), *(const v8h*)(bp + kc + 16), 0, 1, 2, 3, 4, 5, 6, 7, 8, 9, 10, 11, 12, 13, 14, 15);
            if (kc + 64 <= E_NO) {
                __builtin_prefetch(b0 + kc + 32, 0, 1);
                __builtin_prefetch(b1 + kc + 32, 0, 1);
            }
            accE0 = __builtin_amdgcn_wmma_f32_16x16x32_f16(
                false, a0, false, b, (short)0, accE0, false, false);
            accE1 = __builtin_amdgcn_wmma_f32_16x16x32_f16(
                false, a1, false, b, (short)0, accE1, false, false);
        }
        // tail (16 K values, branch-free)
        {
            v16h a0 = load_a_tailE(b0, kc, aoff);
            v16h a1 = load_a_tailE(b1, kc, aoff);
            v16h b  = __builtin_shufflevector(*(const v8h*)(bp + kc), *(const v8h*)(bp + kc + 16), 0, 1, 2, 3, 4, 5, 6, 7, 8, 9, 10, 11, 12, 13, 14, 15);   // padded with zeros
            accE0 = __builtin_amdgcn_wmma_f32_16x16x32_f16(
                false, a0, false, b, (short)0, accE0, false, false);
            accE1 = __builtin_amdgcn_wmma_f32_16x16x32_f16(
                false, a1, false, b, (short)0, accE1, false, false);
        }
    }

    // ---------------- I path: K = 500 ----------------
    {
        const float* b0 = S_i + gRow0 * I_NO;
        const float* b1 = S_i + gRow1 * I_NO;
        const _Float16* bp = Bi + (size_t)row * KPAD_I + boff;
        int kc = 0;
        for (; kc + 32 <= I_NO; kc += 32) {
            v16h a0 = load_a_full(b0, kc, aoff);
            v16h a1 = load_a_full(b1, kc, aoff);
            v16h b  = __builtin_shufflevector(*(const v8h*)(bp + kc), *(const v8h*)(bp + kc + 16), 0, 1, 2, 3, 4, 5, 6, 7, 8, 9, 10, 11, 12, 13, 14, 15);
            accI0 = __builtin_amdgcn_wmma_f32_16x16x32_f16(
                false, a0, false, b, (short)0, accI0, false, false);
            accI1 = __builtin_amdgcn_wmma_f32_16x16x32_f16(
                false, a1, false, b, (short)0, accI1, false, false);
        }
        // tail (20 K values, branch-free via cndmask)
        {
            v16h a0 = load_a_tailI(b0, kc, aoff);
            v16h a1 = load_a_tailI(b1, kc, aoff);
            v16h b  = __builtin_shufflevector(*(const v8h*)(bp + kc), *(const v8h*)(bp + kc + 16), 0, 1, 2, 3, 4, 5, 6, 7, 8, 9, 10, 11, 12, 13, 14, 15);   // padded with zeros
            accI0 = __builtin_amdgcn_wmma_f32_16x16x32_f16(
                false, a0, false, b, (short)0, accI0, false, false);
            accI1 = __builtin_amdgcn_wmma_f32_16x16x32_f16(
                false, a1, false, b, (short)0, accI1, false, false);
        }
    }

    // ---------------- store D (16x16 f32 C/D layout) ----------------
    if (row < SUB) {
#pragma unroll
        for (int r8 = 0; r8 < 8; ++r8) {
            int M = r8 + (hi ? 8 : 0);
            size_t o0 = (size_t)(tile0 * 16 + M) * SUB + row;
            size_t o1 = (size_t)((tile0 + 1) * 16 + M) * SUB + row;
            IN_e[o0] = accE0[r8];
            IN_e[o1] = accE1[r8];
            IN_i[o0] = accI0[r8];
            IN_i[o1] = accI1[r8];
        }
    }
}

// ---------------------------------------------------------------------------
// Kernel 3: causal depthwise conv, 200 taps. One wave per subunit,
// 32 timesteps per block; window staged in LDS with odd row stride.
// ---------------------------------------------------------------------------
__global__ __launch_bounds__(384) void causal_conv_k(
    const float* __restrict__ IN_e, const float* __restrict__ IN_i,
    const float* __restrict__ kS, const float* __restrict__ kNS,
    float* __restrict__ synS, float* __restrict__ synNS)
{
    __shared__ float se[(CONV_T + T_SYN - 1) * CONV_P];
    __shared__ float si[(CONV_T + T_SYN - 1) * CONV_P];

    const int t0 = blockIdx.x * CONV_T;
    const int nrows = CONV_T + T_SYN - 1;          // 231
    for (int i = threadIdx.x; i < nrows * SUB; i += blockDim.x) {
        int r = i / SUB, c = i % SUB;
        int g = t0 - (T_SYN - 1) + r;
        float ve = 0.0f, vi = 0.0f;
        if (g >= 0) {
            ve = IN_e[(size_t)g * SUB + c];
            vi = IN_i[(size_t)g * SUB + c];
        }
        se[r * CONV_P + c] = ve;
        si[r * CONV_P + c] = vi;
    }
    __syncthreads();

    const int s  = threadIdx.x >> 5;   // wave id == subunit (0..11)
    const int tl = threadIdx.x & 31;
    const int t  = t0 + tl;

    const float* kse = kS  + (s * 2 + 0) * T_SYN;
    const float* ksi = kS  + (s * 2 + 1) * T_SYN;
    const float* kne = kNS + (s * 2 + 0) * T_SYN;
    const float* kni = kNS + (s * 2 + 1) * T_SYN;

    float aS = 0.0f, aN = 0.0f;
    for (int k = 0; k < T_SYN; ++k) {
        int lr = tl - k + (T_SYN - 1);
        float xe = se[lr * CONV_P + s];
        float xi = si[lr * CONV_P + s];
        aS = fmaf(kse[k], xe, fmaf(ksi[k], xi, aS));
        aN = fmaf(kne[k], xe, fmaf(kni[k], xi, aN));
    }
    synS [(size_t)t * SUB + s] = aS;
    synNS[(size_t)t * SUB + s] = aN;
}

// ---------------------------------------------------------------------------
// Kernel 4: sequential 20000-step scan; one wave32, lane = subunit,
// C_den @ Y via shuffle broadcasts. (hist/prop terms are identically zero.)
// ---------------------------------------------------------------------------
__global__ __launch_bounds__(32) void scan_k(
    const float* __restrict__ synS, const float* __restrict__ synNS,
    const float* __restrict__ C_den, const float* __restrict__ Theta_s,
    const float* __restrict__ Theta_ns, const float* __restrict__ W_s_sub,
    const float* __restrict__ W_ns_sub, float* __restrict__ out)
{
    const int  s   = threadIdx.x;
    const bool act = (s < SUB);
    const int  sc  = act ? s : 0;      // clamped index: loads unconditional

    float cden[SUB];
#pragma unroll
    for (int j = 0; j < SUB; ++j) {
        float v = C_den[sc * SUB + j];
        cden[j] = act ? v : 0.0f;
    }
    const float th_s  = act ? Theta_s[sc]  : 0.0f;
    const float th_ns = act ? Theta_ns[sc] : 0.0f;
    const float ws    = act ? W_s_sub[sc]  : 0.0f;
    const float wns   = act ? W_ns_sub[sc] : 0.0f;

    float Ys = 0.0f, Yns = 0.0f;
    for (int t = 0; t < T_DATA; ++t) {
        float xs  = synS [(size_t)t * SUB + sc] + th_s;
        float xns = synNS[(size_t)t * SUB + sc] + th_ns;
#pragma unroll
        for (int j = 0; j < SUB; ++j) {
            float ysj  = __shfl(Ys,  j, 32);
            float ynsj = __shfl(Yns, j, 32);
            xs  = fmaf(cden[j], ysj,  xs);
            xns = fmaf(cden[j], ynsj, xns);
        }
        float sig_s  = 1.0f / (1.0f + expf(-xs));
        float sig_ns = 1.0f / (1.0f + expf(-xns));
        float Yns_n = sig_ns * wns;
        float Ys_n  = ((s == 0) ? sig_s : sig_ns) * ws;
        if (act) {
            float* o = out + (size_t)t * (3 * SUB - 1);
            o[s]       = Ys_n;
            o[SUB + s] = Yns_n;
            if (s >= 1) o[2 * SUB + (s - 1)] = sig_s;
        }
        Ys = Ys_n;
        Yns = Yns_n;
    }
}

// ---------------------------------------------------------------------------
extern "C" void kernel_launch(void* const* d_in, const int* in_sizes, int n_in,
                              void* d_out, int out_size, void* d_ws, size_t ws_size,
                              hipStream_t stream)
{
    const float* S_e      = (const float*)d_in[0];
    const float* S_i      = (const float*)d_in[1];
    const float* C_syn_e  = (const float*)d_in[2];
    const float* C_syn_i  = (const float*)d_in[3];
    const float* C_den    = (const float*)d_in[4];
    const float* W_s_syn  = (const float*)d_in[5];
    const float* W_ns_syn = (const float*)d_in[6];
    const float* D_s      = (const float*)d_in[7];
    const float* D_ns     = (const float*)d_in[8];
    const float* Th_s     = (const float*)d_in[9];
    const float* Th_ns    = (const float*)d_in[10];
    const float* W_s_sub  = (const float*)d_in[11];
    const float* W_ns_sub = (const float*)d_in[12];
    // d_in[13..16] (hist/prop weights) contribute exactly zero here.

    float* wsf   = (float*)d_ws;
    float* IN_e  = wsf;
    float* IN_i  = IN_e  + (size_t)T_DATA * SUB;
    float* kS    = IN_i  + (size_t)T_DATA * SUB;
    float* kNS   = kS    + SUB * 2 * T_SYN;
    float* synS  = kNS   + SUB * 2 * T_SYN;
    float* synNS = synS  + (size_t)T_DATA * SUB;
    _Float16* Be = (_Float16*)(synNS + (size_t)T_DATA * SUB);
    _Float16* Bi = Be + 16 * KPAD_E;
    float* out   = (float*)d_out;

    const int nk = SUB * 2 * T_SYN;
    syn_kernels_k<<<(nk + 255) / 256, 256, 0, stream>>>(W_s_syn,  D_s,  kS);
    syn_kernels_k<<<(nk + 255) / 256, 256, 0, stream>>>(W_ns_syn, D_ns, kNS);
    pack_b_k<<<(16 * KPAD_E + 255) / 256, 256, 0, stream>>>(C_syn_e, Be, E_NO, KPAD_E);
    pack_b_k<<<(16 * KPAD_I + 255) / 256, 256, 0, stream>>>(C_syn_i, Bi, I_NO, KPAD_I);

    const int pairs = (T_DATA / 16) / 2;            // 625 wave-pairs
    spike_gemm_k<<<(pairs + 3) / 4, 128, 0, stream>>>(S_e, S_i, Be, Bi,
                                                      IN_e, IN_i);

    causal_conv_k<<<T_DATA / CONV_T, 384, 0, stream>>>(IN_e, IN_i, kS, kNS,
                                                       synS, synNS);

    scan_k<<<1, 32, 0, stream>>>(synS, synNS, C_den, Th_s, Th_ns,
                                 W_s_sub, W_ns_sub, out);
}
